// CSPNet_21053929685602
// MI455X (gfx1250) — matrix operand order repaired and checked by reading, not compiled, not run on hardware
//
#include <hip/hip_runtime.h>

typedef __bf16 bf16_t;
typedef __bf16 v8bf  __attribute__((ext_vector_type(8)));
typedef __bf16 v16bf __attribute__((ext_vector_type(16)));
typedef float  v8f   __attribute__((ext_vector_type(8)));

#define NB 64       // graphs
#define NA 32       // atoms per graph
#define NN 2048     // nodes
#define NE 65536    // edges
#define HH 512      // hidden
#define LLAT 256    // latent
#define NLAYER 4

__device__ inline float silu_f(float x) { return x / (1.0f + __expf(-x)); }

__device__ inline v8f vzero() {
    v8f r;
#pragma unroll
    for (int i = 0; i < 8; ++i) r[i] = 0.0f;
    return r;
}

__device__ inline v8f bwmma(v16bf a, v16bf b, v8f c) {
    return __builtin_amdgcn_wmma_f32_16x16x32_bf16(false, a, false, b, (short)0, c,
                                                   false, false);
}

// A-matrix fragment (16x32 bf16, MxK) from a row-major buffer (LDS or global).
// ISA 7.12.2 layout reduces to two contiguous 8-element runs per lane:
//   e=0..7  -> k = half*8 + e
//   e=8..15 -> k = 16 + half*8 + (e-8)
// -> two aligned 16B loads.
__device__ inline v16bf load_a_frag(const bf16_t* __restrict__ base, int ldk, int lane,
                                    int kt) {
    int m = lane & 15;
    int half = lane >> 4;
    const bf16_t* p = base + (size_t)m * ldk + kt + half * 8;
    v8bf lo = *(const v8bf*)p;
    v8bf hi = *(const v8bf*)(p + 16);
    return __builtin_shufflevector(lo, hi, 0, 1, 2, 3, 4, 5, 6, 7, 8, 9, 10, 11, 12, 13,
                                   14, 15);
}

// B-matrix fragment (32x16 bf16, KxN) from transposed weights Wt[n][k] (row-major, ld=ldk).
// e=0..7 -> k = half*8 + e; e=8..15 -> k = 16 + half*8 + (e-8).  Two aligned 16B loads.
__device__ inline v16bf load_b_frag(const bf16_t* __restrict__ Wt, int ldk, int col0,
                                    int lane, int kt) {
    int n = lane & 15;
    int half = lane >> 4;
    const bf16_t* p = Wt + (size_t)(col0 + n) * ldk + kt + half * 8;
    v8bf lo = *(const v8bf*)p;
    v8bf hi = *(const v8bf*)(p + 16);
    return __builtin_shufflevector(lo, hi, 0, 1, 2, 3, 4, 5, 6, 7, 8, 9, 10, 11, 12, 13,
                                   14, 15);
}

// ---------------- weight convert + transpose: Wt[n][k] = W[k][n], K padded to Kp ---------
__global__ void convtr_k(const float* __restrict__ W, bf16_t* __restrict__ Wt, int K,
                         int N, int Kp, long total) {
    long idx = (long)blockIdx.x * blockDim.x + threadIdx.x;
    if (idx >= total) return;
    int n = (int)(idx / Kp);
    int k = (int)(idx % Kp);
    Wt[idx] = (k < K) ? (bf16_t)W[(size_t)k * N + n] : (bf16_t)0.0f;
}

// ---------------- A0 = concat(emb[atom-1], t[graph])  [N,768] bf16 -----------------------
__global__ void build_a0_k(const float* __restrict__ emb, const float* __restrict__ t,
                           const int* __restrict__ atom, const int* __restrict__ n2g,
                           bf16_t* __restrict__ A0) {
    long idx = (long)blockIdx.x * blockDim.x + threadIdx.x;
    if (idx >= (long)NN * 768) return;
    int nd = (int)(idx / 768);
    int c = (int)(idx % 768);
    float v = (c < HH) ? emb[(size_t)(atom[nd] - 1) * HH + c]
                       : t[(size_t)n2g[nd] * LLAT + (c - HH)];
    A0[idx] = (bf16_t)v;
}

// ---------------- sinusoid edge features: fd[E,64] bf16 (60 used, 4 pad) -----------------
__global__ void fdemb_k(const float* __restrict__ frac, bf16_t* __restrict__ fd) {
    int e = blockIdx.x * blockDim.x + threadIdx.x;
    if (e >= NE) return;
    int g = e >> 10;
    int i = (e >> 5) & 31;
    int j = e & 31;
    int src = g * NA + i, dst = g * NA + j;
    bf16_t* o = fd + (size_t)e * 64;
#pragma unroll
    for (int d = 0; d < 3; ++d) {
        float df = frac[dst * 3 + d] - frac[src * 3 + d];
        df -= floorf(df);
#pragma unroll
        for (int f = 0; f < 10; ++f) {
            float ang = 6.2831853071795864f * (float)f * df;
            o[d * 10 + f] = (bf16_t)__sinf(ang);
            o[30 + d * 10 + f] = (bf16_t)__cosf(ang);
        }
    }
    o[60] = (bf16_t)0.0f; o[61] = (bf16_t)0.0f; o[62] = (bf16_t)0.0f; o[63] = (bf16_t)0.0f;
}

// ---------------- lattice inner products [B,9] -------------------------------------------
__global__ void latip_k(const float* __restrict__ lat, float* __restrict__ lip) {
    int idx = blockIdx.x * blockDim.x + threadIdx.x;
    if (idx >= NB * 9) return;
    int b = idx / 9, i = (idx % 9) / 3, k = idx % 3;
    float s = 0.f;
#pragma unroll
    for (int j = 0; j < 3; ++j) s += lat[b * 9 + i * 3 + j] * lat[b * 9 + k * 3 + j];
    lip[idx] = s;
}

// ---------------- R[g][c] = lat_ip[g] . W1c[:,c]   (W1c: 9x512 f32 slice) ----------------
__global__ void rvec_k(const float* __restrict__ lip, const float* __restrict__ W1c,
                       float* __restrict__ R) {
    int idx = blockIdx.x * blockDim.x + threadIdx.x;
    if (idx >= NB * HH) return;
    int g = idx / HH, c = idx % HH;
    float s = 0.f;
#pragma unroll
    for (int q = 0; q < 9; ++q) s += lip[g * 9 + q] * W1c[(size_t)q * HH + c];
    R[idx] = s;
}

// ---------------- generic WMMA GEMM: C = act(A[M,KP] @ Wt^T + bias) (+resid) -------------
// block 256 = 8 waves, block tile 128x64, wave tile 32x32 (2 M-frags x 2 N-frags).
// Per k-step: 4 WMMAs vs 4 fragment loads (B frags reused across both A frags).
// A-frags straight from global (two 16B loads/lane) -> no barriers in the K loop.
template <int KP, int ACT>
__global__ __launch_bounds__(256) void gemm_k(const bf16_t* __restrict__ A,
                                              const bf16_t* __restrict__ Wt,
                                              const float* __restrict__ bias,
                                              const float* __restrict__ resid,
                                              float* __restrict__ Cf,
                                              bf16_t* __restrict__ Cb) {
    int tid = threadIdx.x;
    int lane = tid & 31, w = tid >> 5;
    int m0 = blockIdx.y * 128 + (w & 3) * 32;
    int n0 = blockIdx.x * 64 + (w >> 2) * 32;
    v8f acc[2][2];
#pragma unroll
    for (int i = 0; i < 2; ++i)
#pragma unroll
        for (int j = 0; j < 2; ++j) acc[i][j] = vzero();
    const bf16_t* Abase = A + (size_t)m0 * KP;

#pragma unroll 4
    for (int kt = 0; kt < KP; kt += 32) {
        v16bf a0 = load_a_frag(Abase, KP, lane, kt);
        v16bf a1 = load_a_frag(Abase + (size_t)16 * KP, KP, lane, kt);
        v16bf b0 = load_b_frag(Wt, KP, n0, lane, kt);
        v16bf b1 = load_b_frag(Wt, KP, n0 + 16, lane, kt);
        acc[0][0] = bwmma(a0, b0, acc[0][0]);
        acc[0][1] = bwmma(a0, b1, acc[0][1]);
        acc[1][0] = bwmma(a1, b0, acc[1][0]);
        acc[1][1] = bwmma(a1, b1, acc[1][1]);
    }

    int nloc = lane & 15, mh = lane >> 4;
#pragma unroll
    for (int nf = 0; nf < 2; ++nf) {
        int col = n0 + nf * 16 + nloc;
        float bc = bias ? bias[col] : 0.0f;
#pragma unroll
        for (int mf = 0; mf < 2; ++mf) {
#pragma unroll
            for (int r = 0; r < 8; ++r) {
                int row = m0 + mf * 16 + mh * 8 + r;
                float x = acc[mf][nf][r] + bc;
                if (ACT) x = silu_f(x);
                if (resid) x += resid[(size_t)row * HH + col];
                if (Cf) Cf[(size_t)row * HH + col] = x;
                if (Cb) Cb[(size_t)row * HH + col] = (bf16_t)x;
            }
        }
    }
}

// ---------------- fused edge MLP1 + MLP2 + scatter-mean ----------------------------------
// one block per src node n: 32 edges (contiguous), h1 kept in LDS, emits agg[n][512]
__global__ __launch_bounds__(256) void edge_fused_k(
    const bf16_t* __restrict__ fd, const bf16_t* __restrict__ W1dt,  // [512,64]
    const bf16_t* __restrict__ W2t,                                  // [512,512]
    const float* __restrict__ Pf, const float* __restrict__ Qf,
    const float* __restrict__ Rf, const float* __restrict__ b1,
    const float* __restrict__ b2, float* __restrict__ agg) {
    __shared__ __align__(16) bf16_t sFd[32 * 64];
    __shared__ __align__(16) bf16_t sH[32 * 512];
    int tid = threadIdx.x, lane = tid & 31, w = tid >> 5;
    int n = blockIdx.x;
    int g = n >> 5;
    int c0 = w * 64;
    int nloc = lane & 15, mh = lane >> 4;

    ((uint4*)sFd)[tid] = ((const uint4*)(fd + (size_t)n * 2048))[tid];
    __syncthreads();

    // phase 1: S = fd @ W1d   (M=32,K=64,N=512; this wave: 64 cols)
    v8f S[2][4];
#pragma unroll
    for (int a = 0; a < 2; ++a)
#pragma unroll
        for (int b = 0; b < 4; ++b) S[a][b] = vzero();
#pragma unroll
    for (int ks = 0; ks < 64; ks += 32) {
        v16bf a0 = load_a_frag(sFd, 64, lane, ks);
        v16bf a1 = load_a_frag(sFd + 16 * 64, 64, lane, ks);
#pragma unroll
        for (int nt = 0; nt < 4; ++nt) {
            v16bf b = load_b_frag(W1dt, 64, c0 + nt * 16, lane, ks);
            S[0][nt] = bwmma(a0, b, S[0][nt]);
            S[1][nt] = bwmma(a1, b, S[1][nt]);
        }
    }
    // h1 = silu(P[n] + Q[dst] + R[g] + S + b1) -> LDS bf16
#pragma unroll
    for (int nt = 0; nt < 4; ++nt) {
        int col = c0 + nt * 16 + nloc;
        float addc = Pf[(size_t)n * HH + col] + Rf[(size_t)g * HH + col] + b1[col];
#pragma unroll
        for (int mt = 0; mt < 2; ++mt)
#pragma unroll
            for (int r = 0; r < 8; ++r) {
                int j = mt * 16 + mh * 8 + r;
                float x = S[mt][nt][r] + addc + Qf[(size_t)(g * NA + j) * HH + col];
                sH[j * 512 + col] = (bf16_t)silu_f(x);
            }
    }
    __syncthreads();

    // phase 2: ef = silu(h1 @ W2 + b2); agg[n] = mean over 32 rows
    v8f C[2][4];
#pragma unroll
    for (int a = 0; a < 2; ++a)
#pragma unroll
        for (int b = 0; b < 4; ++b) C[a][b] = vzero();
#pragma unroll 4
    for (int kt = 0; kt < 512; kt += 32) {
        v16bf a0 = load_a_frag(sH, 512, lane, kt);
        v16bf a1 = load_a_frag(sH + 16 * 512, 512, lane, kt);
#pragma unroll
        for (int nt = 0; nt < 4; ++nt) {
            v16bf b = load_b_frag(W2t, 512, c0 + nt * 16, lane, kt);
            C[0][nt] = bwmma(a0, b, C[0][nt]);
            C[1][nt] = bwmma(a1, b, C[1][nt]);
        }
    }
#pragma unroll
    for (int nt = 0; nt < 4; ++nt) {
        int col = c0 + nt * 16 + nloc;
        float b2c = b2[col];
        float xs = 0.0f;
#pragma unroll
        for (int mt = 0; mt < 2; ++mt)
#pragma unroll
            for (int r = 0; r < 8; ++r) xs += silu_f(C[mt][nt][r] + b2c);
        xs += __shfl_xor(xs, 16);
        if (mh == 0) agg[(size_t)n * HH + col] = xs * (1.0f / 32.0f);
    }
}

// ---------------- nin = concat(nf, agg) bf16 [N,1024] ------------------------------------
__global__ void build_nin_k(const float* __restrict__ nf, const float* __restrict__ agg,
                            bf16_t* __restrict__ nin) {
    long idx = (long)blockIdx.x * blockDim.x + threadIdx.x;
    if (idx >= (long)NN * 1024) return;
    int nd = (int)(idx / 1024);
    int c = (int)(idx % 1024);
    float v = (c < HH) ? nf[(size_t)nd * HH + c] : agg[(size_t)nd * HH + (c - HH)];
    nin[idx] = (bf16_t)v;
}

// ---------------- output heads -----------------------------------------------------------
__global__ void coord_k(const float* __restrict__ nf, const float* __restrict__ cW,
                        float* __restrict__ out) {
    int idx = blockIdx.x * blockDim.x + threadIdx.x;
    if (idx >= NN * 3) return;
    int nd = idx / 3, d = idx % 3;
    float s = 0.f;
    for (int c = 0; c < HH; ++c) s += nf[(size_t)nd * HH + c] * cW[(size_t)c * 3 + d];
    out[NB * 9 + idx] = s;
}

__global__ void gf_k(const float* __restrict__ nf, float* __restrict__ gf) {
    int idx = blockIdx.x * blockDim.x + threadIdx.x;
    if (idx >= NB * HH) return;
    int g = idx / HH, c = idx % HH;
    float s = 0.f;
#pragma unroll 4
    for (int j = 0; j < NA; ++j) s += nf[(size_t)(g * NA + j) * HH + c];
    gf[idx] = s * (1.0f / (float)NA);
}

__global__ void latout_k(const float* __restrict__ gf, const float* __restrict__ lW,
                         const float* __restrict__ lat, float* __restrict__ out) {
    int idx = blockIdx.x * blockDim.x + threadIdx.x;
    if (idx >= NB * 9) return;
    int b = idx / 9, i = (idx % 9) / 3, k = idx % 3;
    float tj[3];
#pragma unroll
    for (int j = 0; j < 3; ++j) {
        float s = 0.f;
        for (int c = 0; c < HH; ++c) s += gf[(size_t)b * HH + c] * lW[(size_t)c * 9 + i * 3 + j];
        tj[j] = s;
    }
    float s = 0.f;
#pragma unroll
    for (int j = 0; j < 3; ++j) s += tj[j] * lat[b * 9 + j * 3 + k];
    out[idx] = s;
}

// =========================================================================================
extern "C" void kernel_launch(void* const* d_in, const int* in_sizes, int n_in,
                              void* d_out, int out_size, void* d_ws, size_t ws_size,
                              hipStream_t stream) {
    (void)in_sizes; (void)n_in; (void)out_size; (void)ws_size;
    const float* t      = (const float*)d_in[0];
    const float* frac   = (const float*)d_in[1];
    const float* latt   = (const float*)d_in[2];
    const float* emb    = (const float*)d_in[3];
    const float* Wlat   = (const float*)d_in[4];
    const float* blat   = (const float*)d_in[5];
    const float* eW1    = (const float*)d_in[6];
    const float* eb1    = (const float*)d_in[7];
    const float* eW2    = (const float*)d_in[8];
    const float* eb2    = (const float*)d_in[9];
    const float* nW1    = (const float*)d_in[10];
    const float* nb1    = (const float*)d_in[11];
    const float* nW2    = (const float*)d_in[12];
    const float* nb2    = (const float*)d_in[13];
    const float* coordW = (const float*)d_in[14];
    const float* latW   = (const float*)d_in[15];
    const int* atom     = (const int*)d_in[16];
    const int* n2g      = (const int*)d_in[17];
    float* out = (float*)d_out;

    // workspace carve-out (256B aligned)
    char* base = (char*)d_ws;
    size_t off = 0;
    auto alloc = [&](size_t bytes) -> void* {
        void* p = base + off;
        off += (bytes + 255) & ~(size_t)255;
        return p;
    };
    bf16_t* wWlat_t = (bf16_t*)alloc((size_t)HH * 768 * 2);
    bf16_t* wW1a_t  = (bf16_t*)alloc((size_t)NLAYER * HH * HH * 2);
    bf16_t* wW1b_t  = (bf16_t*)alloc((size_t)NLAYER * HH * HH * 2);
    bf16_t* wW1d_t  = (bf16_t*)alloc((size_t)NLAYER * HH * 64 * 2);
    bf16_t* wW2_t   = (bf16_t*)alloc((size_t)NLAYER * HH * HH * 2);
    bf16_t* wNW1_t  = (bf16_t*)alloc((size_t)NLAYER * HH * 1024 * 2);
    bf16_t* wNW2_t  = (bf16_t*)alloc((size_t)NLAYER * HH * HH * 2);
    bf16_t* A0      = (bf16_t*)alloc((size_t)NN * 768 * 2);
    float*  nf_f    = (float*)alloc((size_t)NN * HH * 4);
    bf16_t* nf_b    = (bf16_t*)alloc((size_t)NN * HH * 2);
    float*  P_f     = (float*)alloc((size_t)NN * HH * 4);
    float*  Q_f     = (float*)alloc((size_t)NN * HH * 4);
    float*  R_f     = (float*)alloc((size_t)NB * HH * 4);
    float*  lip     = (float*)alloc((size_t)NB * 9 * 4);
    bf16_t* fd_b    = (bf16_t*)alloc((size_t)NE * 64 * 2);
    float*  agg_f   = (float*)alloc((size_t)NN * HH * 4);
    bf16_t* nin_b   = (bf16_t*)alloc((size_t)NN * 1024 * 2);
    bf16_t* u_b     = (bf16_t*)alloc((size_t)NN * HH * 2);
    float*  gf_f    = (float*)alloc((size_t)NB * HH * 4);

    auto conv = [&](const float* W, bf16_t* Wt, int K, int N, int Kp) {
        long total = (long)N * Kp;
        convtr_k<<<(int)((total + 255) / 256), 256, 0, stream>>>(W, Wt, K, N, Kp, total);
    };

    // weight conversion (transposed bf16, K padded where needed)
    conv(Wlat, wWlat_t, 768, HH, 768);
    for (int i = 0; i < NLAYER; ++i) {
        conv(eW1 + ((size_t)i * 1093 + 0) * HH,    wW1a_t + (size_t)i * HH * HH,  HH, HH, HH);
        conv(eW1 + ((size_t)i * 1093 + 512) * HH,  wW1b_t + (size_t)i * HH * HH,  HH, HH, HH);
        conv(eW1 + ((size_t)i * 1093 + 1033) * HH, wW1d_t + (size_t)i * HH * 64,  60, HH, 64);
        conv(eW2 + (size_t)i * HH * HH,            wW2_t  + (size_t)i * HH * HH,  HH, HH, HH);
        conv(nW1 + (size_t)i * 1024 * HH,          wNW1_t + (size_t)i * HH * 1024, 1024, HH, 1024);
        conv(nW2 + (size_t)i * HH * HH,            wNW2_t + (size_t)i * HH * HH,  HH, HH, HH);
    }

    // precompute
    build_a0_k<<<(NN * 768 + 255) / 256, 256, 0, stream>>>(emb, t, atom, n2g, A0);
    fdemb_k<<<NE / 256, 256, 0, stream>>>(frac, fd_b);
    latip_k<<<(NB * 9 + 255) / 256, 256, 0, stream>>>(latt, lip);

    dim3 gg(HH / 64, NN / 128);

    // nf = A0 @ W_latent + b_latent
    gemm_k<768, 0><<<gg, 256, 0, stream>>>(A0, wWlat_t, blat, nullptr, nf_f, nf_b);

    for (int i = 0; i < NLAYER; ++i) {
        gemm_k<512, 0><<<gg, 256, 0, stream>>>(nf_b, wW1a_t + (size_t)i * HH * HH,
                                               nullptr, nullptr, P_f, nullptr);
        gemm_k<512, 0><<<gg, 256, 0, stream>>>(nf_b, wW1b_t + (size_t)i * HH * HH,
                                               nullptr, nullptr, Q_f, nullptr);
        rvec_k<<<(NB * HH + 255) / 256, 256, 0, stream>>>(
            lip, eW1 + ((size_t)i * 1093 + 1024) * HH, R_f);
        edge_fused_k<<<NN, 256, 0, stream>>>(fd_b, wW1d_t + (size_t)i * HH * 64,
                                             wW2_t + (size_t)i * HH * HH, P_f, Q_f, R_f,
                                             eb1 + (size_t)i * HH, eb2 + (size_t)i * HH,
                                             agg_f);
        build_nin_k<<<(NN * 1024 + 255) / 256, 256, 0, stream>>>(nf_f, agg_f, nin_b);
        gemm_k<1024, 1><<<gg, 256, 0, stream>>>(nin_b, wNW1_t + (size_t)i * HH * 1024,
                                                nb1 + (size_t)i * HH, nullptr, nullptr,
                                                u_b);
        gemm_k<512, 1><<<gg, 256, 0, stream>>>(u_b, wNW2_t + (size_t)i * HH * HH,
                                               nb2 + (size_t)i * HH, nf_f, nf_f, nf_b);
    }

    // heads: lattice_out first (576 floats), then coord_out (6144 floats)
    gf_k<<<(NB * HH + 255) / 256, 256, 0, stream>>>(nf_f, gf_f);
    latout_k<<<(NB * 9 + 255) / 256, 256, 0, stream>>>(gf_f, latW, latt, out);
    coord_k<<<(NN * 3 + 255) / 256, 256, 0, stream>>>(nf_f, coordW, out);
}
